// NeuralOdeModel_19731079758396
// MI455X (gfx1250) — matrix operand order repaired and checked by reading, not compiled, not run on hardware
//
#include <hip/hip_runtime.h>
#include <hip/hip_bf16.h>
#include <math.h>

typedef __attribute__((ext_vector_type(16))) __bf16 v16bf;
typedef __attribute__((ext_vector_type(8)))  __bf16 v8bf;
typedef __attribute__((ext_vector_type(8)))  float  v8f;

#define FCH    76            // channels
#define IP     80            // padded channels (M pad, row stride)
#define TT     256           // central T-tile
#define HALO   32            // = receptive field of 32 chained k=3 convs
#define TW     (TT + 2*HALO) // 320
#define NCLS   15
#define TTOT   4096
#define BATCH  64
#define BD     320           // 10 waves
#define NSTEPS 8

#define LDS_BYTES ((TW*IP*4)*2 + (TW*IP*2)*2)   // hs + acc + hinA + hinB = 307200

// Native CDNA5 tanh (TRANS32 op).
__device__ __forceinline__ float fast_tanh(float v) {
#if __has_builtin(__builtin_amdgcn_tanhf)
  return __builtin_amdgcn_tanhf(v);
#else
  float r;
  asm("v_tanh_f32 %0, %1\n\tv_nop" : "=v"(r) : "v"(v));
  return r;
#endif
}

// Software-pipeline hint: keep one dk-group (6 ds_load_b128) in flight ahead
// of each WMMA triplet so waits become partial instead of dscnt==0 drains.
__device__ __forceinline__ void tile_sched_hint() {
#if __has_builtin(__builtin_amdgcn_sched_group_barrier)
  __builtin_amdgcn_sched_group_barrier(0x100, 6, 0); // DS reads: dk0 B frags
  __builtin_amdgcn_sched_group_barrier(0x100, 6, 0); // DS reads: dk1 B frags
  __builtin_amdgcn_sched_group_barrier(0x008, 3, 0); // WMMA: dk0
  __builtin_amdgcn_sched_group_barrier(0x100, 6, 0); // DS reads: dk2 B frags
  __builtin_amdgcn_sched_group_barrier(0x008, 3, 0); // WMMA: dk1
  __builtin_amdgcn_sched_group_barrier(0x008, 3, 0); // WMMA: dk2
#endif
}

__global__ __launch_bounds__(BD, 1)
void node_rk4_kernel(const float* __restrict__ x,
                     const float* __restrict__ conv_w,
                     const float* __restrict__ conv_b,
                     const float* __restrict__ final_w,
                     const float* __restrict__ final_b,
                     float* __restrict__ out)
{
  extern __shared__ char smem[];
  float*  hs   = (float*)smem;            // [TW][IP] f32 state
  float*  acc  = hs + TW*IP;              // [TW][IP] f32 RK4 accumulator
  __bf16* hinA = (__bf16*)(acc + TW*IP);  // [TW][IP] bf16 stage input (dbl buf)
  __bf16* hinB = hinA + TW*IP;

  const int tile = blockIdx.x;
  const int b    = blockIdx.y;
  const int t0   = tile*TT - HALO;        // global t of LDS row 0
  const int tid  = threadIdx.x;
  const int lane = tid & 31;
  const int wave = tid >> 5;
  const int ml   = lane & 15;             // M row within tile / N col within tile
  const int g    = lane >> 4;             // lane group (K split)
  const int mt   = wave % 5;              // this wave's fixed M-tile

  // ---- A fragments: conv weights -> bf16, WMMA 16x16x32 A layout, in regs ----
  // K chunks over in-channels: bases {0,32,48}; chunk2 masks duplicated 48..63.
  const int kbase0 = 0, kbase1 = 32, kbase2 = 48;
  v16bf afrag[3][3];
  const int orow = mt*16 + ml;
  #pragma unroll
  for (int dk = 0; dk < 3; ++dk) {
    #pragma unroll
    for (int kc = 0; kc < 3; ++kc) {
      int kb = (kc == 0) ? kbase0 : (kc == 1 ? kbase1 : kbase2);
      #pragma unroll
      for (int e = 0; e < 16; ++e) {
        // A element e -> K = e + 16*laneGroup   (matches B contiguous layout)
        int K = e + 16*g;
        int i = kb + K;
        float wv = 0.0f;
        if (orow < FCH && i < FCH && !(kc == 2 && i < 64))
          wv = conv_w[(orow*FCH + i)*3 + dk];
        afrag[dk][kc][e] = (__bf16)wv;
      }
    }
  }

  float biasv[8];
  #pragma unroll
  for (int r = 0; r < 8; ++r) {           // D layout: VGPR r -> M = r + 8*g
    int oo = mt*16 + r + 8*g;
    biasv[r] = (oo < FCH) ? conv_b[oo] : 0.0f;
  }

  // ---- initial state load: hs = x (transposed view), masked halo & pad ----
  for (int idx = tid; idx < TW*IP; idx += BD) {
    int t = idx / IP, f = idx - t*IP;
    int tg = t0 + t;
    float v = 0.0f;
    if (f < FCH && tg >= 0 && tg < TTOT)
      v = x[((size_t)b*TTOT + tg)*FCH + f];
    hs[idx]   = v;
    acc[idx]  = 0.0f;
    hinA[idx] = (__bf16)v;
  }
  __syncthreads();

  const float dt = 1.0f / NSTEPS;

  for (int step = 0; step < NSTEPS; ++step) {
    for (int s = 0; s < 4; ++s) {
      const __bf16* hin  = (s & 1) ? hinB : hinA;
      __bf16*       hout = (s & 1) ? hinA : hinB;
      // branchless stage coefficients (pure s_cselect, no constant-mem loads)
      const float cs = (s == 2) ? dt : ((s == 3) ? 0.0f : 0.5f*dt);
      const float wk = (s == 1 || s == 2) ? (dt/3.0f) : (dt/6.0f);

      for (int j = 0; j < 10; ++j) {          // this wave's N-tiles
        int nt = (wave / 5) + 2*j;
        int tl = nt*16 + ml;                  // LDS t row for this lane's column

        // ---- 9 B-fragment loads (18x ds_load_b128) ----
        v16bf bfrag[3][3];
        #pragma unroll
        for (int dk = 0; dk < 3; ++dk) {
          int tp = tl + dk - 1;               // conv shift; clamp only hits garbage rim
          tp = tp < 0 ? 0 : (tp > TW-1 ? TW-1 : tp);
          const __bf16* rp = hin + tp*IP + 16*g;   // B element e -> K = e + 16*g
          #pragma unroll
          for (int kc = 0; kc < 3; ++kc) {
            int kb = (kc == 0) ? kbase0 : (kc == 1 ? kbase1 : kbase2);
            const v8bf* p = (const v8bf*)(rp + kb);   // 16B-aligned ds_load_b128 x2
            v8bf lo = p[0], hi = p[1];
            bfrag[dk][kc] = __builtin_shufflevector(lo, hi,
                0,1,2,3,4,5,6,7,8,9,10,11,12,13,14,15);
          }
        }

        // ---- 9 WMMAs ----
        v8f c = {};
        #pragma unroll
        for (int dk = 0; dk < 3; ++dk)
          #pragma unroll
          for (int kc = 0; kc < 3; ++kc)
            c = __builtin_amdgcn_wmma_f32_16x16x32_bf16(
                  false, afrag[dk][kc], false, bfrag[dk][kc],
                  (short)0, c, false, false);

        tile_sched_hint();

        // ---- epilogue: k = tanh(y+b); acc += w*k; next stage = bf16(h + c*k) ----
        float kv[8];
        #pragma unroll
        for (int r = 0; r < 8; ++r)
          kv[r] = fast_tanh(c[r] + biasv[r]);
        int off = tl*IP + mt*16 + 8*g;        // lane's 8 contiguous channels
        #pragma unroll
        for (int r = 0; r < 8; ++r)
          acc[off + r] += wk * kv[r];
        if (s < 3) {
          int tg = t0 + tl;
          bool valid = (tg >= 0) && (tg < TTOT);  // emulate conv zero-padding
          v8bf hv;
          #pragma unroll
          for (int r = 0; r < 8; ++r)
            hv[r] = valid ? (__bf16)(hs[off + r] + cs*kv[r]) : (__bf16)0.0f;
          *(v8bf*)(hout + off) = hv;
        }
      }
      __syncthreads();
    }
    // h += RK4 increment; reset acc; rebuild masked bf16 state for next k1
    for (int idx = tid; idx < TW*IP; idx += BD) {
      float v = hs[idx] + acc[idx];
      hs[idx]  = v;
      acc[idx] = 0.0f;
      int t = idx / IP;
      int tg = t0 + t;
      hinA[idx] = (tg >= 0 && tg < TTOT) ? (__bf16)v : (__bf16)0.0f;
    }
    __syncthreads();
  }

  // ---- final linear 76 -> 15 on central 256 columns (reuse acc LDS) ----
  float* fw = acc;                 // [NCLS][FCH]
  float* fb = acc + NCLS*FCH;
  for (int idx = tid; idx < NCLS*FCH; idx += BD) fw[idx] = final_w[idx];
  if (tid < NCLS) fb[tid] = final_b[tid];
  __syncthreads();

  for (int t = tid; t < TT; t += BD) {
    const float* hr = hs + (HALO + t)*IP;
    size_t obase = ((size_t)b*TTOT + (size_t)tile*TT + t)*NCLS;
    #pragma unroll 1
    for (int c = 0; c < NCLS; ++c) {
      float sum = fb[c];
      const float* wr = fw + c*FCH;
      for (int f = 0; f < FCH; ++f)
        sum += hr[f]*wr[f];
      out[obase + c] = sum;
    }
  }
}

extern "C" void kernel_launch(void* const* d_in, const int* in_sizes, int n_in,
                              void* d_out, int out_size, void* d_ws, size_t ws_size,
                              hipStream_t stream) {
  const float* x       = (const float*)d_in[0];
  const float* conv_w  = (const float*)d_in[1];
  const float* conv_b  = (const float*)d_in[2];
  const float* final_w = (const float*)d_in[3];
  const float* final_b = (const float*)d_in[4];
  float* out = (float*)d_out;

  // Raise dynamic-LDS cap to CDNA5's 320KB/WGP (host-side attr, graph-safe).
  (void)hipFuncSetAttribute((const void*)node_rk4_kernel,
                            hipFuncAttributeMaxDynamicSharedMemorySize,
                            LDS_BYTES);

  dim3 grid(TTOT/TT, BATCH);   // 16 x 64 = 1024 blocks, one WGP each (LDS-bound)
  dim3 block(BD);
  hipLaunchKernelGGL(node_rk4_kernel, grid, block, LDS_BYTES, stream,
                     x, conv_w, conv_b, final_w, final_b, out);
}
